// ResBlock_66073776881955
// MI455X (gfx1250) — matrix operand order repaired and checked by reading, not compiled, not run on hardware
//
#include <hip/hip_runtime.h>

// ---------------------------------------------------------------------------
// Submanifold sparse conv ResBlock on MI455X (gfx1250, wave32).
//   conv1 -> BN+ReLU -> conv2 -> BN+ReLU -> residual add
// bf16 V_WMMA_F32_16X16X32_BF16 (16384 FLOP/instr), fp32 accumulation.
//
// Each wave owns FOUR 16-row tiles (64 rows x 64 cols of output,
// 16x v8f accumulators). Per kernel offset k:
//   - 16x global_load_b128 gathers build the A-fragments directly in the
//     ISA 16-bit A 16x32 layout (no LDS, no shuffles)
//   - 8x 32B B-fragment loads from a pre-swizzled L2-resident weight buffer,
//     each immediately feeding 4 back-to-back WMMAs (4x B-traffic reuse:
//     ~1 GB/conv from L2 instead of 4 GB with one tile/wave)
// BN batch stats fused in the epilogue: cross-half-wave shuffle reduce +
// one fp32 atomic pair per column per 64 rows.
// ---------------------------------------------------------------------------

#define KOFF 27
#define CH   64
#define TPW  4        // 16-row tiles per wave

typedef __attribute__((ext_vector_type(16))) __bf16 v16bf;
typedef __attribute__((ext_vector_type(8)))  __bf16 v8bf;
typedef __attribute__((ext_vector_type(8)))  float  v8f;

union AFrag { v16bf v; v8bf h[2]; };

// ---- fp32 -> bf16 elementwise convert ------------------------------------
__global__ void cvt_bf16_kernel(const float* __restrict__ src,
                                __bf16* __restrict__ dst, int total) {
  int i = blockIdx.x * blockDim.x + threadIdx.x;
  if (i < total) dst[i] = (__bf16)src[i];
}

// ---- pre-swizzle W[27][64][64] into WMMA B-fragment order -----------------
// dest layout: frag block ((k*2+h)*4+t) of 512 bf16; within a block,
// lane*16 + el, where element el of lane l is
//   B[h*32 + (el<8 ? (l>=16?8:0) + el : 16 + (l>=16?8:0) + el - 8)][t*16 + (l&15)]
// (16-bit B 32x16 layout: lane n, VGPR j = {K=2j,2j+1} (j<4), {K=2j+8,2j+9} (j>=4), col n; lanes 16-31 K+8)
__global__ void prep_w_kernel(const float* __restrict__ W,
                              __bf16* __restrict__ Wb) {
  int tid = blockIdx.x * blockDim.x + threadIdx.x;
  const int TOT = KOFF * 2 * 4 * 512;
  if (tid >= TOT) return;
  int blk  = tid >> 9;          // ((k*2+h)*4+t)
  int e    = tid & 511;
  int lane = e >> 4;
  int el   = e & 15;
  int t = blk & 3;
  int h = (blk >> 2) & 1;
  int k = blk >> 3;
  int kk  = h * 32 + ((el < 8) ? ((lane >= 16) ? 8 : 0) + el : 16 + ((lane >= 16) ? 8 : 0) + el - 8);   // input channel
  int col = t * 16 + (lane & 15);                    // output channel
  Wb[tid] = (__bf16)W[(k * CH + kk) * CH + col];
}

// ---- the WMMA conv: one wave = four 16-row tiles (64 rows x 64 cols) ------
__global__ __launch_bounds__(256)
void subm_conv_wmma_kernel(const __bf16* __restrict__ X,
                           const __bf16* __restrict__ Wb,
                           const int*    __restrict__ nbr,
                           float*        __restrict__ out,
                           float*        __restrict__ sum,
                           float*        __restrict__ sumsq,
                           int Nrows) {
  const int lane      = threadIdx.x & 31;
  const int wave      = threadIdx.x >> 5;
  const int group     = blockIdx.x * 8 + wave;       // 64-row group
  const int numTiles  = (Nrows + 15) >> 4;
  const int numGroups = (numTiles + TPW - 1) / TPW;
  if (group >= numGroups) return;

  const int tile0  = group * TPW;
  const int laneLo = lane & 15;
  const int hi     = lane >> 4;        // half-wave select
  const int aoff   = hi ? 8 : 0;       // A-frag sub-offset within a 32-ch half

  v8f acc[TPW][4];
#pragma unroll
  for (int tt = 0; tt < TPW; ++tt)
#pragma unroll
    for (int t = 0; t < 4; ++t)
#pragma unroll
      for (int i = 0; i < 8; ++i) acc[tt][t][i] = 0.0f;

  int rows[TPW];
#pragma unroll
  for (int tt = 0; tt < TPW; ++tt) rows[tt] = (tile0 + tt) * 16 + laneLo;

  for (int k = 0; k < KOFF; ++k) {
    // ---- gather A-fragments for all 4 tiles (both K-halves) ---------------
    AFrag a[TPW][2];
#pragma unroll
    for (int tt = 0; tt < TPW; ++tt) {
      int nb = (rows[tt] < Nrows) ? nbr[rows[tt] * KOFF + k] : -1;
      if (nb >= 0) {
        const __bf16* base = X + (size_t)nb * CH;
        // 16-bit A 16x32 layout -> two contiguous 16B chunks per K-half:
        //  lane<16: K = {0..7, 16..23};  lane>=16: K = {8..15, 24..31}
        a[tt][0].h[0] = *(const v8bf*)(base + aoff);
        a[tt][0].h[1] = *(const v8bf*)(base + 16 + aoff);
        a[tt][1].h[0] = *(const v8bf*)(base + 32 + aoff);
        a[tt][1].h[1] = *(const v8bf*)(base + 48 + aoff);
      } else {
#pragma unroll
        for (int i = 0; i < 8; ++i) {
          a[tt][0].h[0][i] = (__bf16)0.0f; a[tt][0].h[1][i] = (__bf16)0.0f;
          a[tt][1].h[0][i] = (__bf16)0.0f; a[tt][1].h[1][i] = (__bf16)0.0f;
        }
      }
    }

    // ---- 8 B-fragment loads, each feeding 4 WMMAs -------------------------
    const __bf16* wk = Wb + (size_t)k * (2 * 4 * 512) + lane * 16;
#pragma unroll
    for (int h = 0; h < 2; ++h)
#pragma unroll
      for (int t = 0; t < 4; ++t) {
        v16bf b = *(const v16bf*)(wk + (h * 4 + t) * 512);
#pragma unroll
        for (int tt = 0; tt < TPW; ++tt)
          acc[tt][t] = __builtin_amdgcn_wmma_f32_16x16x32_bf16(
              false, a[tt][h].v, false, b, (short)0, acc[tt][t], false, false);
      }
  }

  // ---- store D (32-bit C/D layout: VGPR i -> M = i + 8*hi, N = laneLo) ----
#pragma unroll
  for (int tt = 0; tt < TPW; ++tt)
#pragma unroll
    for (int t = 0; t < 4; ++t)
#pragma unroll
      for (int i = 0; i < 8; ++i) {
        int row = (tile0 + tt) * 16 + i + 8 * hi;
        if (row < Nrows) out[(size_t)row * CH + t * 16 + laneLo] = acc[tt][t][i];
      }

  // ---- fused BN statistics: per-column sum / sumsq over these 64 rows -----
  // (out-of-range rows contributed exact zeros, so no guard needed)
#pragma unroll
  for (int t = 0; t < 4; ++t) {
    float s = 0.0f, q = 0.0f;
#pragma unroll
    for (int tt = 0; tt < TPW; ++tt)
#pragma unroll
      for (int i = 0; i < 8; ++i) { float v = acc[tt][t][i]; s += v; q += v * v; }
    s += __shfl_xor(s, 16, 32);        // combine the two half-waves
    q += __shfl_xor(q, 16, 32);
    if (hi == 0) {
      atomicAdd(&sum[t * 16 + laneLo], s);
      atomicAdd(&sumsq[t * 16 + laneLo], q);
    }
  }
}

// ---- BN finalize: mu/var -> scale/shift -----------------------------------
__global__ void bn_finalize_kernel(const float* __restrict__ sum,
                                   const float* __restrict__ sumsq,
                                   const float* __restrict__ gamma,
                                   const float* __restrict__ beta,
                                   float* __restrict__ scale,
                                   float* __restrict__ shift, float invN) {
  int c = threadIdx.x;
  float mu  = sum[c] * invN;
  float var = sumsq[c] * invN - mu * mu;
  float sc  = gamma[c] * rsqrtf(var + 1e-5f);
  scale[c] = sc;
  shift[c] = beta[c] - mu * sc;
}

// ---- BN+ReLU after conv1: emit fp32 x_net and bf16 copy for conv2 ---------
__global__ void bn_relu_mid_kernel(const float* __restrict__ conv,
                                   const float* __restrict__ scale,
                                   const float* __restrict__ shift,
                                   float* __restrict__ xf,
                                   __bf16* __restrict__ xb, int total) {
  int i = blockIdx.x * blockDim.x + threadIdx.x;
  if (i >= total) return;
  int c = i & (CH - 1);
  float v = fmaxf(conv[i] * scale[c] + shift[c], 0.0f);
  xf[i] = v;
  xb[i] = (__bf16)v;
}

// ---- BN+ReLU after conv2 + residual add -----------------------------------
__global__ void final_kernel(const float* __restrict__ conv,
                             const float* __restrict__ xf,
                             const float* __restrict__ scale,
                             const float* __restrict__ shift,
                             float* __restrict__ out, int total) {
  int i = blockIdx.x * blockDim.x + threadIdx.x;
  if (i >= total) return;
  int c = i & (CH - 1);
  out[i] = xf[i] + fmaxf(conv[i] * scale[c] + shift[c], 0.0f);
}

// ---------------------------------------------------------------------------
extern "C" void kernel_launch(void* const* d_in, const int* in_sizes, int n_in,
                              void* d_out, int out_size, void* d_ws, size_t ws_size,
                              hipStream_t stream) {
  const float* feats  = (const float*)d_in[0];
  const float* W1     = (const float*)d_in[1];
  const float* gamma1 = (const float*)d_in[2];
  const float* beta1  = (const float*)d_in[3];
  const float* W2     = (const float*)d_in[4];
  const float* gamma2 = (const float*)d_in[5];
  const float* beta2  = (const float*)d_in[6];
  const int*   nbr    = (const int*)d_in[7];

  const int N     = in_sizes[0] / CH;   // active voxels
  const int total = N * CH;

  // ---- carve workspace ----------------------------------------------------
  char* ws = (char*)d_ws;
  size_t off = 0;
  auto carve = [&](size_t bytes) -> void* {
    void* p = ws + off;
    off += (bytes + 255) & ~(size_t)255;
    return p;
  };
  const size_t WB_ELEMS = (size_t)KOFF * 2 * 4 * 512;   // 110592 bf16
  __bf16* featsB = (__bf16*)carve((size_t)total * 2);
  __bf16* xnetB  = (__bf16*)carve((size_t)total * 2);
  __bf16* W1b    = (__bf16*)carve(WB_ELEMS * 2);
  __bf16* W2b    = (__bf16*)carve(WB_ELEMS * 2);
  float*  convO  = (float*)carve((size_t)total * 4);    // reused by both convs
  float*  xnetF  = (float*)carve((size_t)total * 4);
  float*  stats  = (float*)carve(4 * CH * sizeof(float));
  float *sum = stats, *sumsq = stats + CH, *scale = stats + 2 * CH, *shift = stats + 3 * CH;

  const int thr = 256;
  const int tiles      = (N + 15) / 16;
  const int groups     = (tiles + TPW - 1) / TPW;       // 64-row groups
  const int convBlocks = (groups + 7) / 8;              // 8 waves per block
  const float invN = 1.0f / (float)N;

  // ---- prep: bf16 feats + swizzled bf16 weights ---------------------------
  cvt_bf16_kernel<<<(total + thr - 1) / thr, thr, 0, stream>>>(feats, featsB, total);
  prep_w_kernel<<<((int)WB_ELEMS + thr - 1) / thr, thr, 0, stream>>>(W1, W1b);
  prep_w_kernel<<<((int)WB_ELEMS + thr - 1) / thr, thr, 0, stream>>>(W2, W2b);

  // ---- conv1 -> BN+ReLU ---------------------------------------------------
  hipMemsetAsync(sum, 0, 2 * CH * sizeof(float), stream);
  subm_conv_wmma_kernel<<<convBlocks, thr, 0, stream>>>(featsB, W1b, nbr, convO, sum, sumsq, N);
  bn_finalize_kernel<<<1, CH, 0, stream>>>(sum, sumsq, gamma1, beta1, scale, shift, invN);
  bn_relu_mid_kernel<<<(total + thr - 1) / thr, thr, 0, stream>>>(convO, scale, shift, xnetF, xnetB, total);

  // ---- conv2 -> BN+ReLU -> residual add -----------------------------------
  hipMemsetAsync(sum, 0, 2 * CH * sizeof(float), stream);
  subm_conv_wmma_kernel<<<convBlocks, thr, 0, stream>>>(xnetB, W2b, nbr, convO, sum, sumsq, N);
  bn_finalize_kernel<<<1, CH, 0, stream>>>(sum, sumsq, gamma2, beta2, scale, shift, invN);
  final_kernel<<<(total + thr - 1) / thr, thr, 0, stream>>>(convO, xnetF, scale, shift, (float*)d_out, total);
}
